// Attention_2138893714255
// MI455X (gfx1250) — matrix operand order repaired and checked by reading, not compiled, not run on hardware
//
#include <hip/hip_runtime.h>
#include <math.h>

typedef __bf16 bf16;
typedef __attribute__((ext_vector_type(16))) __bf16 v16bf;
typedef __attribute__((ext_vector_type(8)))  __bf16 v8bf;
typedef __attribute__((ext_vector_type(4)))  __bf16 v4bf;
typedef __attribute__((ext_vector_type(8)))  float   v8f;

static constexpr int kB  = 4;
static constexpr int kT  = 2048;
static constexpr int kD  = 1024;
static constexpr int kNH = 8;
static constexpr int kHD = 128;
static constexpr int kM  = kB * kT;         // 8192 rows
static constexpr float kScale = 0.08838834764831845f; // 1/sqrt(128)
static constexpr float kCap = 30.0f;

// ---------------- WMMA helpers ----------------
__device__ __forceinline__ v8f v8f_zero() {
  v8f z = {0.f,0.f,0.f,0.f,0.f,0.f,0.f,0.f};
  return z;
}

__device__ __forceinline__ v8f wmma_bf16(v16bf a, v16bf b, v8f c) {
  return __builtin_amdgcn_wmma_f32_16x16x32_bf16(false, a, false, b, (short)0, c, false, false);
}

// A fragment (16x32, row-major source). lane: m = l&15, h = l>>4.
// element e: K = (e&7) + 8*h + 16*(e>>3) -> two contiguous 16B runs.
__device__ __forceinline__ v16bf ld_a16(const bf16* __restrict__ row, int k0, int h) {
  v8bf lo = *(const v8bf*)(row + k0 + 8*h);
  v8bf hi = *(const v8bf*)(row + k0 + 16 + 8*h);
  return __builtin_shufflevector(lo, hi, 0,1,2,3,4,5,6,7,8,9,10,11,12,13,14,15);
}

// B fragment (32x16). Source is N-major (row n holds K contiguously).
// lane: n = l&15, h = l>>4. element e: K = (e&7) + 8*h + 16*(e>>3) -> two contiguous 16B runs.
__device__ __forceinline__ v16bf ld_b16(const bf16* __restrict__ rowN, int k0, int h) {
  v8bf lo = *(const v8bf*)(rowN + k0 + 8*h);
  v8bf hi = *(const v8bf*)(rowN + k0 + 16 + 8*h);
  return __builtin_shufflevector(lo, hi, 0,1,2,3,4,5,6,7,8,9,10,11,12,13,14,15);
}

// ---------------- Kernel 1: f32 -> bf16 convert ----------------
__global__ void cvt_f32_to_bf16(const float4* __restrict__ in, v4bf* __restrict__ out, int n4) {
  int i = blockIdx.x * blockDim.x + threadIdx.x;
  if (i < n4) {
    float4 v = in[i];
    v4bf o = { (bf16)v.x, (bf16)v.y, (bf16)v.z, (bf16)v.w };
    out[i] = o;
  }
}

// ---------------- Kernel 2: fused QKV projection + RoPE + gain ----------------
// Units per 16-row m-tile: 32 Q column-pairs (8 heads x 4), 4 K pairs, 4 V pairs.
// A pair is columns (c, c+64) within a head so RoPE rotates inside one wave.
__global__ void qkv_rope_kernel(const bf16* __restrict__ xb,
                                const bf16* __restrict__ wq,
                                const bf16* __restrict__ wk,
                                const bf16* __restrict__ wv,
                                const float* __restrict__ gain,
                                const float* __restrict__ cosT,
                                const float* __restrict__ sinT,
                                bf16* __restrict__ qo,   // [B][NH][T][HD]
                                bf16* __restrict__ ko,   // [B][T][HD]
                                bf16* __restrict__ vto)  // [B][HD][T]
{
  int wid   = blockIdx.x * (blockDim.x >> 5) + (threadIdx.x >> 5);
  int lane  = threadIdx.x & 31;
  int l15   = lane & 15;
  int h     = lane >> 4;
  int mtile = wid / 40;
  int u     = wid - mtile * 40;

  const bf16* W;
  int cbase;        // base column of first tile inside this weight's output space
  int p;            // pair index within the 64-wide half
  int hq = 0;
  float g = 1.0f;
  if (u < 32)      { hq = u >> 2; p = u & 3; W = wq; g = gain[hq]; cbase = hq*kHD + p*16; }
  else if (u < 36) { p = u - 32;  W = wk;    cbase = p*16; }
  else             { p = u - 36;  W = wv;    cbase = p*16; }

  int n0 = cbase + l15;
  int n1 = n0 + 64;
  const bf16* arow  = xb + (size_t)(mtile*16 + l15) * kD;
  const bf16* b0row = W + (size_t)n0 * kD;
  const bf16* b1row = W + (size_t)n1 * kD;

  v8f acc0 = v8f_zero();
  v8f acc1 = v8f_zero();
  for (int k0 = 0; k0 < kD; k0 += 32) {
    v16bf a  = ld_a16(arow, k0, h);
    v16bf b0 = ld_b16(b0row, k0, h);
    v16bf b1 = ld_b16(b1row, k0, h);
    acc0 = wmma_bf16(a, b0, acc0);
    acc1 = wmma_bf16(a, b1, acc1);
  }

  int mb = mtile * 16;
  int b  = mb >> 11;            // T = 2048 = 2^11
  if (u < 36) {
    // RoPE: x1 = col j (first half), x2 = col j+64; angle index j = p*16 + l15
    int jf = p*16 + l15;
    #pragma unroll
    for (int r = 0; r < 8; ++r) {
      int   mg = mb + r + 8*h;
      int   t  = mg & (kT - 1);
      float c  = cosT[t*64 + jf];
      float s  = sinT[t*64 + jf];
      float x1 = acc0[r], x2 = acc1[r];
      float o1 = (x1*c - x2*s) * g;
      float o2 = (x1*s + x2*c) * g;
      if (u < 32) {
        size_t base = ((size_t)(b*kNH + hq)*kT + t) * kHD;
        qo[base + p*16 + l15]      = (bf16)o1;
        qo[base + p*16 + l15 + 64] = (bf16)o2;
      } else {
        size_t base = ((size_t)b*kT + t) * kHD;
        ko[base + p*16 + l15]      = (bf16)o1;
        ko[base + p*16 + l15 + 64] = (bf16)o2;
      }
    }
  } else {
    // V: store transposed [b][hd][t]; lane owns one column over 8 consecutive rows
    int tb = (mb + 8*h) & (kT - 1);
    v8bf pk0, pk1;
    #pragma unroll
    for (int r = 0; r < 8; ++r) { pk0[r] = (bf16)acc0[r]; pk1[r] = (bf16)acc1[r]; }
    *(v8bf*)(vto + ((size_t)b*kHD + n0) * kT + tb) = pk0;
    *(v8bf*)(vto + ((size_t)b*kHD + n1) * kT + tb) = pk1;
  }
}

// ---------------- Kernel 3: flash attention ----------------
// Block = 256 threads = 8 waves = the 8 heads of one (batch, q-tile).
// Each wave: 16 q-rows x full HD; online softmax over 32-key blocks.
__global__ void __launch_bounds__(256, 1)
flash_kernel(const bf16* __restrict__ q,    // [B][NH][T][HD]
             const bf16* __restrict__ k,    // [B][T][HD]
             const bf16* __restrict__ vt,   // [B][HD][T]
             const float* /*unused*/,
             bf16* __restrict__ attn)       // [B][T][NH*HD]
{
  __shared__ __align__(16) bf16 plds[8][16*32];
  int w    = threadIdx.x >> 5;          // wave == head
  int lane = threadIdx.x & 31;
  int l15  = lane & 15;
  int h    = lane >> 4;
  int qt   = blockIdx.x & 127;          // q-tile within sequence
  int b    = blockIdx.x >> 7;
  int hh   = w;
  int m0   = qt * 16;

  const bf16* qrow = q + ((size_t)(b*kNH + hh)*kT + m0 + l15) * kHD;
  v16bf aq[4];
  #pragma unroll
  for (int i = 0; i < 4; ++i) aq[i] = ld_a16(qrow, i*32, h);

  v8f accO[8];
  float mrow[8], lrow[8];
  #pragma unroll
  for (int i = 0; i < 8; ++i) { accO[i] = v8f_zero(); mrow[i] = -3.0e38f; lrow[i] = 0.f; }

  int kbend = (m0 + 15) >> 5;  // uniform across the block (same qt for all waves)
  for (int kb = 0; kb <= kbend; ++kb) {
    int key0 = kb * 32;
    v8f s0 = v8f_zero(), s1 = v8f_zero();
    const bf16* k0row = k + ((size_t)b*kT + key0 + l15) * kHD;
    const bf16* k1row = k0row + (size_t)16 * kHD;
    #pragma unroll
    for (int i = 0; i < 4; ++i) {
      s0 = wmma_bf16(aq[i], ld_b16(k0row, i*32, h), s0);
      s1 = wmma_bf16(aq[i], ld_b16(k1row, i*32, h), s1);
    }

    float p0[8], p1[8], msc[8];
    #pragma unroll
    for (int r = 0; r < 8; ++r) {
      int   mg = m0 + r + 8*h;
      float x0 = kCap * tanhf(s0[r] * (kScale / kCap));
      float x1 = kCap * tanhf(s1[r] * (kScale / kCap));
      if (key0 + l15 > mg)      x0 = -3.0e38f;
      if (key0 + 16 + l15 > mg) x1 = -3.0e38f;
      float vmax = fmaxf(x0, x1);
      #pragma unroll
      for (int off = 1; off < 16; off <<= 1)
        vmax = fmaxf(vmax, __shfl_xor(vmax, off, 32));
      float mnew = fmaxf(mrow[r], vmax);
      float sc   = __expf(mrow[r] - mnew);
      float e0   = __expf(x0 - mnew);
      float e1   = __expf(x1 - mnew);
      float psum = e0 + e1;
      #pragma unroll
      for (int off = 1; off < 16; off <<= 1)
        psum += __shfl_xor(psum, off, 32);
      lrow[r] = lrow[r] * sc + psum;
      mrow[r] = mnew;
      msc[r]  = sc;
      p0[r] = e0; p1[r] = e1;
    }

    #pragma unroll
    for (int j = 0; j < 8; ++j)
      #pragma unroll
      for (int r = 0; r < 8; ++r) accO[j][r] *= msc[r];

    // transpose P (C/D layout -> A layout) through per-wave LDS tile
    #pragma unroll
    for (int r = 0; r < 8; ++r) {
      int m = r + 8*h;
      plds[w][m*32 + l15]      = (bf16)p0[r];
      plds[w][m*32 + l15 + 16] = (bf16)p1[r];
    }
    __syncthreads();
    v16bf ap = ld_a16(&plds[w][l15 * 32], 0, h);
    __syncthreads();

    #pragma unroll
    for (int j = 0; j < 8; ++j) {
      const bf16* vrow = vt + ((size_t)b*kHD + j*16 + l15) * kT;
      accO[j] = wmma_bf16(ap, ld_b16(vrow, key0, h), accO[j]);
    }
  }

  float inv[8];
  #pragma unroll
  for (int r = 0; r < 8; ++r) inv[r] = 1.0f / lrow[r];
  #pragma unroll
  for (int j = 0; j < 8; ++j)
    #pragma unroll
    for (int r = 0; r < 8; ++r) {
      int t = m0 + r + 8*h;
      attn[((size_t)(b*kT + t))*kD + hh*kHD + j*16 + l15] = (bf16)(accO[j][r] * inv[r]);
    }
}

// ---------------- Kernel 4: output projection ----------------
// One wave -> 16x64 output tile (1 A fragment reused across 4 WMMAs per K step).
__global__ void outproj_kernel(const bf16* __restrict__ attn,  // [M][D]
                               const bf16* __restrict__ wo,    // [D][D] (N-major)
                               float* __restrict__ out)        // [M][D] f32
{
  int wid   = blockIdx.x * (blockDim.x >> 5) + (threadIdx.x >> 5);
  int lane  = threadIdx.x & 31;
  int l15   = lane & 15;
  int h     = lane >> 4;
  int nt    = wid & 15;     // 16 groups of 64 columns
  int mtile = wid >> 4;     // 512 m-tiles

  const bf16* arow = attn + (size_t)(mtile*16 + l15) * kD;
  const bf16* brow[4];
  #pragma unroll
  for (int j = 0; j < 4; ++j) brow[j] = wo + (size_t)(nt*64 + j*16 + l15) * kD;

  v8f acc[4];
  #pragma unroll
  for (int j = 0; j < 4; ++j) acc[j] = v8f_zero();

  for (int k0 = 0; k0 < kD; k0 += 32) {
    v16bf a = ld_a16(arow, k0, h);
    #pragma unroll
    for (int j = 0; j < 4; ++j)
      acc[j] = wmma_bf16(a, ld_b16(brow[j], k0, h), acc[j]);
  }

  #pragma unroll
  for (int j = 0; j < 4; ++j)
    #pragma unroll
    for (int r = 0; r < 8; ++r)
      out[(size_t)(mtile*16 + r + 8*h)*kD + nt*64 + j*16 + l15] = acc[j][r];
}

// ---------------- Host launcher ----------------
extern "C" void kernel_launch(void* const* d_in, const int* in_sizes, int n_in,
                              void* d_out, int out_size, void* d_ws, size_t ws_size,
                              hipStream_t stream) {
  const float* x    = (const float*)d_in[0];
  const float* Wq   = (const float*)d_in[1];
  const float* Wk   = (const float*)d_in[2];
  const float* Wv   = (const float*)d_in[3];
  const float* Wo   = (const float*)d_in[4];
  const float* gain = (const float*)d_in[5];
  const float* cosT = (const float*)d_in[6];
  const float* sinT = (const float*)d_in[7];
  float* out = (float*)d_out;

  char* w = (char*)d_ws;
  size_t off = 0;
  auto carve = [&](size_t elems) { void* p = w + off; off += elems * sizeof(bf16); return (bf16*)p; };
  bf16* xb   = carve((size_t)kM * kD);            // 16 MB
  bf16* wqb  = carve((size_t)kD * kD);            //  2 MB
  bf16* wkb  = carve((size_t)kHD * kD);           // 256 KB
  bf16* wvb  = carve((size_t)kHD * kD);           // 256 KB
  bf16* wob  = carve((size_t)kD * kD);            //  2 MB
  bf16* qb   = carve((size_t)kB * kNH * kT * kHD);// 16 MB
  bf16* kb   = carve((size_t)kB * kT * kHD);      //  2 MB
  bf16* vtb  = carve((size_t)kB * kHD * kT);      //  2 MB
  bf16* attn = carve((size_t)kM * kD);            // 16 MB

  auto cvt = [&](const float* src, bf16* dst, size_t n) {
    int n4 = (int)(n / 4);
    cvt_f32_to_bf16<<<(n4 + 255) / 256, 256, 0, stream>>>(
        (const float4*)src, (v4bf*)dst, n4);
  };
  cvt(x,  xb,  (size_t)kM * kD);
  cvt(Wq, wqb, (size_t)kD * kD);
  cvt(Wk, wkb, (size_t)kHD * kD);
  cvt(Wv, wvb, (size_t)kHD * kD);
  cvt(Wo, wob, (size_t)kD * kD);

  // 512 m-tiles * 40 units = 20480 waves; 4 waves/block
  qkv_rope_kernel<<<5120, 128, 0, stream>>>(xb, wqb, wkb, wvb, gain, cosT, sinT,
                                            qb, kb, vtb);

  // 512 blocks = (b, q-tile); 8 waves/block = heads
  flash_kernel<<<512, 256, 0, stream>>>(qb, kb, vtb, gain, attn);

  // 512 m-tiles * 16 n-groups = 8192 waves; 4 waves/block
  outproj_kernel<<<2048, 128, 0, stream>>>(attn, wob, out);
}
